// LSTM_71502615544558
// MI455X (gfx1250) — compile-verified, matrix operand order checked by reading, not run on hardware
//
#include <hip/hip_runtime.h>

// ---------------------------------------------------------------------------
// 4-layer LSTM (H=128, B=256, T=2048, IN=2) + FC(128->1) for gfx1250.
//
// Wavefront-pipelined over layers: one launch of 64 blocks
// (layer = blockIdx>>4, batch-chunk = blockIdx&15). Each block runs the full
// sequential scan for its 16 batch rows of one layer, with WMMA-f16 gates
// (f32 accum), weights cached in VGPRs as B-fragments, h/x staged in LDS,
// cell state in registers. Layer l consumes layer l-1's f16 [T,B,H] stream
// via agent-scope acquire/release progress counters (deadlock-free: deps
// point only to lower block IDs, which HW dispatches first).
// ---------------------------------------------------------------------------

typedef __attribute__((ext_vector_type(16))) _Float16 v16h;
typedef __attribute__((ext_vector_type(8)))  _Float16 v8h;
typedef __attribute__((ext_vector_type(8)))  float    v8f;

#define HH     128        // hidden size
#define G4     512        // 4*H gates
#define BB     256        // batch
#define TT     2048       // time steps
#define BCHUNK 16         // batch rows per workgroup
#define NTPW   4          // 16-col N-tiles per wave -> 64 gate cols/wave
#define AH_STRIDE 144     // padded row stride (halves) for A-operand LDS
#define GW_STRIDE 516     // padded row stride (floats) for gate LDS
#define NCHUNK (BB / BCHUNK)   // 16
#define NLAYER 4

__device__ __forceinline__ float sigmoid_fast(float x) {
    return 1.0f / (1.0f + __expf(-x));
}
__device__ __forceinline__ float tanh_fast(float x) {
    return 2.0f / (1.0f + __expf(-2.0f * x)) - 1.0f;
}

__device__ __forceinline__ int wait_prog(const int* p, int need, int seen) {
    while (seen < need) {
        seen = __hip_atomic_load(p, __ATOMIC_ACQUIRE, __HIP_MEMORY_SCOPE_AGENT);
        if (seen < need) __builtin_amdgcn_s_sleep(1);
    }
    return seen;
}

struct FusedArgs {
    const float* x0;            // [B][T][2]
    const float* w_ih[NLAYER];  // [G4][din]
    const float* w_hh[NLAYER];  // [G4][H]
    const float* b_ih[NLAYER];  // [G4]
    const float* b_hh[NLAYER];  // [G4]
    _Float16*    buf[3];        // layer 0..2 outputs, f16 [T][B][H]
    _Float16*    hlast;         // layer 3 final h, f16 [B][H]
    int*         prog;          // [3][NCHUNK] progress counters
};

// KSTEPS = 4: layer 0 (input dim 2 folded into elementwise stage)
// KSTEPS = 8: hidden layers (input projection fused into WMMA K as ks 4..7)
// LASTONLY: store only the final timestep's h (layer 3)
template <int KSTEPS, bool LASTONLY>
__device__ __forceinline__ void lstm_scan_body(
    const float*    __restrict__ x0,
    const _Float16* __restrict__ xin,
    const float*    __restrict__ w_ih,
    const float*    __restrict__ w_hh,
    const float*    __restrict__ b_ih,
    const float*    __restrict__ b_hh,
    _Float16*       __restrict__ hout,
    const int* prog_in, int* prog_out, int b0,
    _Float16* s_h, _Float16* s_x, float* s_g, float* s_bias, float* s_wih0)
{
    const int tid  = threadIdx.x;
    const int wave = tid >> 5;
    const int lane = tid & 31;
    const int r    = lane & 15;    // row (A) / col (B) within 16-tile
    const int half = lane >> 4;    // K-half selector per fragment layout

    // ---- Preload B-fragments (weights) into registers, f32 -> f16.
    // 16-bit B (32x16): lane l holds col n=(l&15); elements j:0..7 ->
    // K = 32*ks + (l>>4)*8 + j ; j:8..15 -> K = +16.
    v16h bfrag[NTPW][KSTEPS];
#pragma unroll
    for (int nt = 0; nt < NTPW; ++nt) {
        const int n = (wave * NTPW + nt) * 16 + r;
#pragma unroll
        for (int ks = 0; ks < KSTEPS; ++ks) {
            const float* wrow = (ks < 4) ? (w_hh + (size_t)n * HH)
                                         : (w_ih + (size_t)n * HH);
            const int kb = (ks & 3) * 32 + half * 8;
            v16h f;
#pragma unroll
            for (int j = 0; j < 8; ++j) f[j]     = (_Float16)wrow[kb + j];
#pragma unroll
            for (int j = 0; j < 8; ++j) f[8 + j] = (_Float16)wrow[kb + 16 + j];
            bfrag[nt][ks] = f;
        }
    }

    // ---- One-time LDS setup
    for (int i = tid; i < G4; i += 256) s_bias[i] = b_ih[i] + b_hh[i];
    if (KSTEPS == 4) {
        for (int i = tid; i < G4 * 2; i += 256) s_wih0[i] = w_ih[i];
    }
    for (int i = tid; i < BCHUNK * AH_STRIDE; i += 256) s_h[i] = (_Float16)0.0f;

    // Elementwise ownership: thread -> (batch row eb, 8 hidden cols from ej0)
    const int eb  = tid >> 4;
    const int ej0 = (tid & 15) * 8;

    float c[8];
#pragma unroll
    for (int j = 0; j < 8; ++j) c[j] = 0.0f;

    int seen = 0;  // last observed producer progress
    if (KSTEPS == 8) {
        seen = wait_prog(prog_in, 1, seen);     // x_0 available
        const _Float16* src = xin + ((size_t)0 * BB + b0 + eb) * HH + ej0;
        *(v8h*)&s_x[eb * AH_STRIDE + ej0] = *(const v8h*)src;
    }
    __syncthreads();

    for (int t = 0; t < TT; ++t) {
        // ---------------- WMMA stage: gates = [h | x_t] @ Wcat^T ----------
        v8f acc[NTPW] = {};
#pragma unroll
        for (int ks = 0; ks < KSTEPS; ++ks) {
            const _Float16* abase = (ks < 4) ? s_h : s_x;
            const int kb = (ks & 3) * 32 + half * 8;
            const _Float16* p = abase + r * AH_STRIDE + kb;
            union { v16h v; v8h h2[2]; } a;
            a.h2[0] = *(const v8h*)(p);        // K = kb .. kb+7
            a.h2[1] = *(const v8h*)(p + 16);   // K = kb+16 .. kb+23
#pragma unroll
            for (int nt = 0; nt < NTPW; ++nt) {
                acc[nt] = __builtin_amdgcn_wmma_f32_16x16x32_f16(
                    false, a.v, false, bfrag[nt][ks],
                    (short)0, acc[nt], false, false);
            }
        }
        // Scatter accumulators: D layout: VGPR v, lane l -> M=v+8*(l>>4), N=(l&15).
#pragma unroll
        for (int nt = 0; nt < NTPW; ++nt) {
            const int n = (wave * NTPW + nt) * 16 + r;
#pragma unroll
            for (int v = 0; v < 8; ++v) {
                s_g[(v + 8 * half) * GW_STRIDE + n] = acc[nt][v];
            }
        }
        __syncthreads();

        // ---------------- Elementwise stage -------------------------------
        float xv0 = 0.0f, xv1 = 0.0f;
        if (KSTEPS == 4) {
            const float* xp = x0 + ((size_t)(b0 + eb) * TT + t) * 2;
            xv0 = xp[0];
            xv1 = xp[1];
        }
        v8h hv;
#pragma unroll
        for (int j = 0; j < 8; ++j) {
            const int col = ej0 + j;
            float gi = s_g[eb * GW_STRIDE + col      ] + s_bias[col      ];
            float gf = s_g[eb * GW_STRIDE + col + 128] + s_bias[col + 128];
            float gg = s_g[eb * GW_STRIDE + col + 256] + s_bias[col + 256];
            float go = s_g[eb * GW_STRIDE + col + 384] + s_bias[col + 384];
            if (KSTEPS == 4) {
                gi += s_wih0[(col      ) * 2] * xv0 + s_wih0[(col      ) * 2 + 1] * xv1;
                gf += s_wih0[(col + 128) * 2] * xv0 + s_wih0[(col + 128) * 2 + 1] * xv1;
                gg += s_wih0[(col + 256) * 2] * xv0 + s_wih0[(col + 256) * 2 + 1] * xv1;
                go += s_wih0[(col + 384) * 2] * xv0 + s_wih0[(col + 384) * 2 + 1] * xv1;
            }
            const float ig = sigmoid_fast(gi);
            const float fg = sigmoid_fast(gf);
            const float gt = tanh_fast(gg);
            const float og = sigmoid_fast(go);
            c[j] = fg * c[j] + ig * gt;
            hv[j] = (_Float16)(og * tanh_fast(c[j]));
        }
        // New h -> LDS (next step's A operand) and layer output.
        *(v8h*)&s_h[eb * AH_STRIDE + ej0] = hv;
        if (!LASTONLY) {
            _Float16* op = hout + ((size_t)t * BB + b0 + eb) * HH + ej0;
            *(v8h*)op = hv;
        } else if (t == TT - 1) {
            _Float16* op = hout + (size_t)(b0 + eb) * HH + ej0;
            *(v8h*)op = hv;
        }
        // Prefetch next x_t (after producer has published step t+1).
        if (KSTEPS == 8 && (t + 1) < TT) {
            seen = wait_prog(prog_in, t + 2, seen);
            const _Float16* src = xin + ((size_t)(t + 1) * BB + b0 + eb) * HH + ej0;
            *(v8h*)&s_x[eb * AH_STRIDE + ej0] = *(const v8h*)src;
        }
        // Publish progress every 4 steps (amortizes the release fence).
        const bool sig = (prog_out != nullptr) && (((t & 3) == 3) || (t == TT - 1));
        if (sig) __threadfence();
        __syncthreads();
        if (sig && tid == 0) {
            __hip_atomic_store(prog_out, t + 1, __ATOMIC_RELEASE,
                               __HIP_MEMORY_SCOPE_AGENT);
        }
    }
}

__global__ __launch_bounds__(256)
void lstm_fused_kernel(FusedArgs a)
{
    __shared__ __align__(16) _Float16 s_h[BCHUNK * AH_STRIDE];
    __shared__ __align__(16) _Float16 s_x[BCHUNK * AH_STRIDE];
    __shared__ __align__(16) float    s_g[BCHUNK * GW_STRIDE];
    __shared__ float s_bias[G4];
    __shared__ float s_wih0[G4 * 2];

    const int layer = blockIdx.x >> 4;     // 0..3
    const int chunk = blockIdx.x & (NCHUNK - 1);
    const int b0    = chunk * BCHUNK;

    if (layer == 0) {
        lstm_scan_body<4, false>(a.x0, nullptr,
                                 a.w_ih[0], a.w_hh[0], a.b_ih[0], a.b_hh[0],
                                 a.buf[0], nullptr, &a.prog[0 * NCHUNK + chunk],
                                 b0, s_h, s_x, s_g, s_bias, s_wih0);
    } else if (layer < 3) {
        lstm_scan_body<8, false>(nullptr, a.buf[layer - 1],
                                 a.w_ih[layer], a.w_hh[layer],
                                 a.b_ih[layer], a.b_hh[layer],
                                 a.buf[layer],
                                 &a.prog[(layer - 1) * NCHUNK + chunk],
                                 &a.prog[layer * NCHUNK + chunk],
                                 b0, s_h, s_x, s_g, s_bias, s_wih0);
    } else {
        lstm_scan_body<8, true>(nullptr, a.buf[2],
                                a.w_ih[3], a.w_hh[3], a.b_ih[3], a.b_hh[3],
                                a.hlast,
                                &a.prog[2 * NCHUNK + chunk], nullptr,
                                b0, s_h, s_x, s_g, s_bias, s_wih0);
    }
}

__global__ void init_prog_kernel(int* prog)
{
    if (threadIdx.x < 3 * NCHUNK) prog[threadIdx.x] = 0;
}

__global__ __launch_bounds__(256)
void fc_kernel(const _Float16* __restrict__ hlast, // [B][H]
               const float*    __restrict__ fc_w,  // [1][H]
               const float*    __restrict__ fc_b,  // [1]
               float*          __restrict__ out)   // [B]
{
    const int b = blockIdx.x * blockDim.x + threadIdx.x;
    if (b < BB) {
        const _Float16* hp = hlast + (size_t)b * HH;
        float s = 0.0f;
#pragma unroll 8
        for (int k = 0; k < HH; ++k) s += (float)hp[k] * fc_w[k];
        out[b] = s + fc_b[0];
    }
}

extern "C" void kernel_launch(void* const* d_in, const int* in_sizes, int n_in,
                              void* d_out, int out_size, void* d_ws, size_t ws_size,
                              hipStream_t stream) {
    (void)in_sizes; (void)n_in; (void)out_size; (void)ws_size;

    FusedArgs a;
    a.x0 = (const float*)d_in[0];
    for (int l = 0; l < NLAYER; ++l) {
        a.w_ih[l] = (const float*)d_in[1 + 4 * l];
        a.w_hh[l] = (const float*)d_in[2 + 4 * l];
        a.b_ih[l] = (const float*)d_in[3 + 4 * l];
        a.b_hh[l] = (const float*)d_in[4 + 4 * l];
    }
    const float* fc_w = (const float*)d_in[17];
    const float* fc_b = (const float*)d_in[18];

    // Workspace: 3 x f16 [T][B][H] streams (128 MiB each) + final h + counters.
    const size_t actElems = (size_t)TT * BB * HH;
    _Float16* base = (_Float16*)d_ws;
    a.buf[0] = base;
    a.buf[1] = base + actElems;
    a.buf[2] = base + 2 * actElems;
    a.hlast  = base + 3 * actElems;
    a.prog   = (int*)(base + 3 * actElems + (size_t)BB * HH);

    init_prog_kernel<<<dim3(1), dim3(64), 0, stream>>>(a.prog);
    lstm_fused_kernel<<<dim3(NLAYER * NCHUNK), dim3(256), 0, stream>>>(a);
    fc_kernel<<<dim3(1), dim3(256), 0, stream>>>(a.hlast, fc_w, fc_b, (float*)d_out);
}
